// Eva_62818191671438
// MI455X (gfx1250) — matrix operand order repaired and checked by reading, not compiled, not run on hardware
//
#include <hip/hip_runtime.h>
#include <hip/hip_bf16.h>

typedef __attribute__((ext_vector_type(16))) _Float16 v16h;
typedef __attribute__((ext_vector_type(8)))  _Float16 v8h;
typedef __attribute__((ext_vector_type(8)))  float    v8f;
typedef __attribute__((ext_vector_type(4)))  float    v4f;
typedef __attribute__((ext_vector_type(4)))  int      v4i;

#define B_ 4
#define N_ 1026
#define C_ 1024
#define H_ 16
#define T_ 2
#define D_ 64
#define S_ 1024
#define M_ (B_*N_)      // 4104 rows
#define L_ (S_+1)       // 1025 attention rows per (b,h,t)

#if __has_builtin(__builtin_amdgcn_global_load_async_to_lds_b128)
#define HAVE_ASYNC_LDS 1
#else
#define HAVE_ASYNC_LDS 0
#endif

static __device__ __forceinline__ void wait_async0() {
#if __has_builtin(__builtin_amdgcn_s_wait_asynccnt)
  __builtin_amdgcn_s_wait_asynccnt(0);
#else
  asm volatile("s_wait_asynccnt 0x0" ::: "memory");
#endif
}

static __device__ __forceinline__ v16h cat8(v8h lo, v8h hi) {
  return __builtin_shufflevector(lo, hi, 0,1,2,3,4,5,6,7,8,9,10,11,12,13,14,15);
}
static __device__ __forceinline__ v8f wmma_f16(v16h a, v16h b, v8f c) {
  // D = A(16x32 f16) * B(32x16 f16) + C(16x16 f32)
  return __builtin_amdgcn_wmma_f32_16x16x32_f16(false, a, false, b, (short)0, c, false, false);
}

// DPP row_ror butterfly reductions within each 16-lane row (VALU only, no LDS).
template<int CTRL>
static __device__ __forceinline__ float dpp_f(float x) {
  return __builtin_bit_cast(float,
      __builtin_amdgcn_update_dpp(0, __builtin_bit_cast(int, x), CTRL, 0xF, 0xF, true));
}
static __device__ __forceinline__ float rowmax16(float x) {
  x = fmaxf(x, dpp_f<0x121>(x));  // row_ror:1
  x = fmaxf(x, dpp_f<0x122>(x));  // row_ror:2
  x = fmaxf(x, dpp_f<0x124>(x));  // row_ror:4
  x = fmaxf(x, dpp_f<0x128>(x));  // row_ror:8
  return x;
}
static __device__ __forceinline__ float rowsum16(float x) {
  x += dpp_f<0x121>(x);
  x += dpp_f<0x122>(x);
  x += dpp_f<0x124>(x);
  x += dpp_f<0x128>(x);
  return x;
}

// ---------------------------------------------------------------------------
// GEMM: Y[m,n] = sum_k A[m,k] * W[n,k] (+bias). 64x64 tile per 256-thread WG.
// MODE 0: QKV epilogue -> f16 q/k/v buffers laid out [B,H,N,D], with qkv bias.
//         Q additionally pre-scaled by D^-0.5 = 0.125 (commutes with RoPE and
//         the QK^T dot product; removes per-chunk scaling from attention).
// MODE 1: proj epilogue -> f32 out[m*Nout+n] + proj_b.
// ---------------------------------------------------------------------------
template<int MODE>
__global__ void gemm_k(const float* __restrict__ A, const float* __restrict__ W,
                       const float* __restrict__ bq, const float* __restrict__ bk,
                       const float* __restrict__ bv,
                       _Float16* __restrict__ qo, _Float16* __restrict__ ko,
                       _Float16* __restrict__ vo,
                       float* __restrict__ out, int M, int K, int Nout)
{
  __shared__ _Float16 lx[64*40];   // A tile: 64 m-rows x 32 k (stride 40 halves = 80B)
  __shared__ _Float16 lw[64*40];   // W tile: 64 n-rows x 32 k

  const int tid  = threadIdx.x;
  const int lane = tid & 31;
  const int wid  = tid >> 5;
  const int wm   = wid & 3;        // m sub-tile (16 rows)
  const int wn   = wid >> 2;       // n pair (covers 32 cols)
  const int mBase = blockIdx.y * 64;
  const int nBase = blockIdx.x * 64;

  const int ldrow = tid >> 2;      // 0..63 staging row
  const int koff  = (tid & 3) * 8; // 0,8,16,24

  const int mlan  = lane & 15;
  const int grp   = lane >> 4;
  const int kb    = grp * 8;       // unused; A and B frags both use khalf
  const int khalf = grp * 16;      // B-frag k half offset

  v8f acc0 = {}; v8f acc1 = {};

  for (int kc = 0; kc < K; kc += 32) {
    // ---- stage A tile (f32 -> f16), vectorized b128 loads ----
    {
      const int gm = mBase + ldrow;
      v8h xh;
      if (gm < M) {
        const float* s = A + (size_t)gm * K + kc + koff;
        v4f f0 = *(const v4f*)s;
        v4f f1 = *(const v4f*)(s + 4);
        #pragma unroll
        for (int i = 0; i < 4; ++i) { xh[i] = (_Float16)f0[i]; xh[4+i] = (_Float16)f1[i]; }
        if (kc + 32 < K) __builtin_prefetch(s + 32, 0, 3);
      } else {
        #pragma unroll
        for (int i = 0; i < 8; ++i) xh[i] = (_Float16)0.f;
      }
      *(v8h*)&lx[ldrow*40 + koff] = xh;
    }
    // ---- stage W tile (Nout % 64 == 0, always in range) ----
    {
      const int gn = nBase + ldrow;
      const float* s = W + (size_t)gn * K + kc + koff;
      v4f f0 = *(const v4f*)s;
      v4f f1 = *(const v4f*)(s + 4);
      v8h wh;
      #pragma unroll
      for (int i = 0; i < 4; ++i) { wh[i] = (_Float16)f0[i]; wh[4+i] = (_Float16)f1[i]; }
      if (kc + 32 < K) __builtin_prefetch(s + 32, 0, 3);
      *(v8h*)&lw[ldrow*40 + koff] = wh;
    }
    __syncthreads();

    // ---- fragments + WMMA ----
    const int am = wm*16 + mlan;
    v16h a = cat8(*(const v8h*)&lx[am*40 + khalf],
                  *(const v8h*)&lx[am*40 + khalf + 8]);
    const int bn0 = wn*32 + mlan;
    v16h b0 = cat8(*(const v8h*)&lw[bn0*40 + khalf],
                   *(const v8h*)&lw[bn0*40 + khalf + 8]);
    const int bn1 = bn0 + 16;
    v16h b1 = cat8(*(const v8h*)&lw[bn1*40 + khalf],
                   *(const v8h*)&lw[bn1*40 + khalf + 8]);
    acc0 = wmma_f16(a, b0, acc0);
    acc1 = wmma_f16(a, b1, acc1);
    __syncthreads();
  }

  // ---- epilogue ----
  #pragma unroll
  for (int s = 0; s < 2; ++s) {
    v8f c = s ? acc1 : acc0;
    const int n = nBase + wn*32 + s*16 + mlan;
    #pragma unroll
    for (int vi = 0; vi < 8; ++vi) {
      const int m = mBase + wm*16 + vi + 8*grp;
      if (m >= M) continue;
      float val = c[vi];
      if (MODE == 0) {
        const int sel  = n >> 10;
        const int cidx = n & 1023;
        const float* bias = (sel == 0) ? bq : (sel == 1) ? bk : bv;
        val += bias[cidx];
        if (sel == 0) val *= 0.125f;   // fold softmax scale into Q
        const int h = cidx >> 6, d = cidx & 63;
        const int b = m / N_, pos = m - b * N_;
        _Float16* dst = (sel == 0) ? qo : (sel == 1) ? ko : vo;
        dst[(((size_t)b*H_ + h)*N_ + pos)*D_ + d] = (_Float16)val;
      } else {
        val += bq[n];
        out[(size_t)m * Nout + n] = val;
      }
    }
  }
}

// ---------------------------------------------------------------------------
// RoPE on spatial positions of q/k (in-place, f16).
// ---------------------------------------------------------------------------
__global__ void rope_k(_Float16* __restrict__ q, _Float16* __restrict__ k,
                       const float* __restrict__ rope)
{
  const int idx = blockIdx.x * blockDim.x + threadIdx.x;  // B*H*S*32
  const int i  = idx & 31;
  const int s  = (idx >> 5) & (S_ - 1);
  const int bh = idx >> 15;
  const int pos = T_ + s;
  const size_t base = ((size_t)bh * N_ + pos) * D_;
  const float sn0 = rope[s*128 + 2*i],      sn1 = rope[s*128 + 2*i + 1];
  const float cs0 = rope[s*128 + 64 + 2*i], cs1 = rope[s*128 + 64 + 2*i + 1];
  {
    float x0 = (float)q[base + 2*i], x1 = (float)q[base + 2*i + 1];
    q[base + 2*i]     = (_Float16)(x0*cs0 - x1*sn0);
    q[base + 2*i + 1] = (_Float16)(x1*cs1 + x0*sn1);
  }
  {
    float x0 = (float)k[base + 2*i], x1 = (float)k[base + 2*i + 1];
    k[base + 2*i]     = (_Float16)(x0*cs0 - x1*sn0);
    k[base + 2*i + 1] = (_Float16)(x1*cs1 + x0*sn1);
  }
}

// ---------------------------------------------------------------------------
// Flash attention per (b,h,t): L=1025 rows [token, spatial...], head dim 64.
// Workgroup = 8 waves, 128 query rows; key chunks of 32; online softmax.
// K tile staged with async global->LDS copies (ASYNCcnt) when available.
// Q is pre-scaled by 0.125 so logits need no scaling; only the last chunk
// (keys 1024..1055) needs masking.
// ---------------------------------------------------------------------------
__global__ void attn_k(const _Float16* __restrict__ q, const _Float16* __restrict__ k,
                       const _Float16* __restrict__ v, float* __restrict__ o)
{
  __shared__ _Float16 Kt[32*72];      // [key_local][d], stride 72 halves (144B)
  __shared__ _Float16 Vt[64*40];      // [d][key_local], stride 40 halves (80B)
  __shared__ _Float16 Pl[8][16*40];   // per-wave P tile 16x32, stride 40

  const int tid  = threadIdx.x;
  const int lane = tid & 31;
  const int wid  = tid >> 5;
  const int blk  = blockIdx.x;
  const int qb   = blk % 9;           // 9 q-blocks of 128 cover 1025 rows
  const int bht  = blk / 9;           // (b*H+h)*T + t
  const int t    = bht % T_;
  const int bh   = bht / T_;
  const int rbase = qb*128 + wid*16;

  const int mlan  = lane & 15;
  const int grp   = lane >> 4;
  const int kb    = grp * 8;
  const int khalf = grp * 16;

  // Q fragments (row r -> pos: r==0 is the token row, else spatial T+r-1)
  int r = rbase + mlan; if (r > S_) r = S_;           // clamp; masked at store
  const int qpos = (r == 0) ? t : (T_ + r - 1);
  const _Float16* qrow = q + ((size_t)bh * N_ + qpos) * D_;
  const v16h aq0 = cat8(*(const v8h*)&qrow[khalf],      *(const v8h*)&qrow[khalf + 8]);
  const v16h aq1 = cat8(*(const v8h*)&qrow[32 + khalf], *(const v8h*)&qrow[32 + khalf + 8]);

  v8f oacc[4] = {};
  float mrow[8], lsum[8];
  #pragma unroll
  for (int vi = 0; vi < 8; ++vi) { mrow[vi] = -1e30f; lsum[vi] = 0.f; }

  const int lkey = tid >> 3;          // 0..31 staging key
  const int ldo  = (tid & 7) * 8;     // 0..56 staging d offset

  auto chunk = [&](const int keybase, const bool tail) {
    // ---- stage K tile (async DMA to LDS) and transposed V tile ----
    {
      const int kr = keybase + lkey;
      _Float16* lk = &Kt[lkey*72 + ldo];
      if (kr <= S_) {
        const int kpos = (kr == 0) ? t : (T_ + kr - 1);
        const _Float16* gk = &k[((size_t)bh*N_ + kpos)*D_ + ldo];
#if HAVE_ASYNC_LDS
        __builtin_amdgcn_global_load_async_to_lds_b128(
            (__attribute__((address_space(1))) v4i*)gk,
            (__attribute__((address_space(3))) v4i*)lk, 0, 0);
#else
        *(v8h*)lk = *(const v8h*)gk;
#endif
        v8h vv = *(const v8h*)&v[((size_t)bh*N_ + kpos)*D_ + ldo];
        #pragma unroll
        for (int i = 0; i < 8; ++i) Vt[(ldo + i)*40 + lkey] = vv[i];
      } else {
        v8h z;
        #pragma unroll
        for (int i = 0; i < 8; ++i) z[i] = (_Float16)0.f;
        *(v8h*)lk = z;
        #pragma unroll
        for (int i = 0; i < 8; ++i) Vt[(ldo + i)*40 + lkey] = (_Float16)0.f;
      }
    }
#if HAVE_ASYNC_LDS
    wait_async0();
#endif
    __syncthreads();

    // ---- logits = Q @ K^T for two 16-key subtiles ----
    v8f c0 = {}, c1 = {};
    {
      const int kr0 = mlan;
      v16h b00 = cat8(*(const v8h*)&Kt[kr0*72 + khalf],      *(const v8h*)&Kt[kr0*72 + khalf + 8]);
      v16h b01 = cat8(*(const v8h*)&Kt[kr0*72 + 32 + khalf], *(const v8h*)&Kt[kr0*72 + 32 + khalf + 8]);
      c0 = wmma_f16(aq0, b00, c0);
      c0 = wmma_f16(aq1, b01, c0);
      const int kr1 = 16 + mlan;
      v16h b10 = cat8(*(const v8h*)&Kt[kr1*72 + khalf],      *(const v8h*)&Kt[kr1*72 + khalf + 8]);
      v16h b11 = cat8(*(const v8h*)&Kt[kr1*72 + 32 + khalf], *(const v8h*)&Kt[kr1*72 + 32 + khalf + 8]);
      c1 = wmma_f16(aq0, b10, c1);
      c1 = wmma_f16(aq1, b11, c1);
    }
    const bool in0 = !tail || (keybase + mlan)      <= S_;
    const bool in1 = !tail || (keybase + 16 + mlan) <= S_;

    // ---- online softmax (row = vi + 8*grp, cols striped over 16 lanes) ----
    float alpha[8];
    #pragma unroll
    for (int vi = 0; vi < 8; ++vi) {
      const float x0 = tail ? (in0 ? c0[vi] : -1e30f) : c0[vi];
      const float x1 = tail ? (in1 ? c1[vi] : -1e30f) : c1[vi];
      const float mx = rowmax16(fmaxf(x0, x1));           // DPP row reduction
      const float mnew = fmaxf(mrow[vi], mx);
      alpha[vi] = __expf(mrow[vi] - mnew);
      mrow[vi]  = mnew;
      const float p0 = __expf(x0 - mnew);
      const float p1 = __expf(x1 - mnew);
      const float rs = rowsum16(p0 + p1);                 // DPP row reduction
      lsum[vi] = lsum[vi]*alpha[vi] + rs;
      const int mr = vi + 8*grp;
      Pl[wid][mr*40 + mlan]      = (_Float16)p0;
      Pl[wid][mr*40 + 16 + mlan] = (_Float16)p1;
    }
    __builtin_amdgcn_wave_barrier();   // keep P stores ahead of P loads in schedule

    // ---- O = O*alpha + P @ V ----
    v16h ap = cat8(*(const v8h*)&Pl[wid][mlan*40 + khalf],
                   *(const v8h*)&Pl[wid][mlan*40 + khalf + 8]);
    #pragma unroll
    for (int vi = 0; vi < 8; ++vi) {
      oacc[0][vi] *= alpha[vi]; oacc[1][vi] *= alpha[vi];
      oacc[2][vi] *= alpha[vi]; oacc[3][vi] *= alpha[vi];
    }
    #pragma unroll
    for (int dt = 0; dt < 4; ++dt) {
      const int drow = dt*16 + mlan;
      v16h bv = cat8(*(const v8h*)&Vt[drow*40 + khalf],
                     *(const v8h*)&Vt[drow*40 + khalf + 8]);
      oacc[dt] = wmma_f16(ap, bv, oacc[dt]);
    }
    __syncthreads();
  };

  for (int kc = 0; kc < 32; ++kc) chunk(kc * 32, false);  // keys 0..1023, no mask
  chunk(32 * 32, true);                                   // key 1024 + masked tail

  // ---- finalize and store o[bht][row][d] ----
  #pragma unroll
  for (int vi = 0; vi < 8; ++vi) {
    const int rr = rbase + vi + 8*grp;
    if (rr > S_) continue;
    const float inv = 1.f / lsum[vi];
    float* orow = o + ((size_t)bht * L_ + rr) * D_;
    orow[ 0 + mlan] = oacc[0][vi] * inv;
    orow[16 + mlan] = oacc[1][vi] * inv;
    orow[32 + mlan] = oacc[2][vi] * inv;
    orow[48 + mlan] = oacc[3][vi] * inv;
  }
}

// ---------------------------------------------------------------------------
// Combine: ctx[b, pos, h*64+d] = token row (pos<T) or mean over t of spatial row.
// ---------------------------------------------------------------------------
__global__ void combine_k(const float* __restrict__ o, float* __restrict__ ctx)
{
  const int idx = blockIdx.x * blockDim.x + threadIdx.x;
  if (idx >= B_*N_*C_) return;
  const int c   = idx & (C_ - 1);
  const int pos = (idx >> 10) % N_;
  const int b   = idx / (C_ * N_);
  const int h = c >> 6, d = c & 63;
  const int bh = b * H_ + h;
  float val;
  if (pos < T_) {
    val = o[(((size_t)bh*T_ + pos)*L_ + 0)*D_ + d];
  } else {
    const int rr = 1 + (pos - T_);
    val = 0.5f * (o[(((size_t)bh*T_ + 0)*L_ + rr)*D_ + d] +
                  o[(((size_t)bh*T_ + 1)*L_ + rr)*D_ + d]);
  }
  ctx[idx] = val;
}

// ---------------------------------------------------------------------------
extern "C" void kernel_launch(void* const* d_in, const int* in_sizes, int n_in,
                              void* d_out, int out_size, void* d_ws, size_t ws_size,
                              hipStream_t stream)
{
  (void)in_sizes; (void)n_in; (void)out_size; (void)ws_size;
  const float* x      = (const float*)d_in[0];
  const float* rope   = (const float*)d_in[1];
  const float* qkv_w  = (const float*)d_in[2];
  const float* q_bias = (const float*)d_in[3];
  const float* k_bias = (const float*)d_in[4];
  const float* v_bias = (const float*)d_in[5];
  const float* proj_w = (const float*)d_in[6];
  const float* proj_b = (const float*)d_in[7];
  float* out = (float*)d_out;

  char* ws = (char*)d_ws;
  const size_t qkvElems = (size_t)B_ * H_ * N_ * D_;              // 4,202,496
  _Float16* qb = (_Float16*)ws;
  _Float16* kb = qb + qkvElems;
  _Float16* vb = kb + qkvElems;
  float* ob  = (float*)(ws + 3 * qkvElems * sizeof(_Float16));    // B*H*T*L*D f32
  float* ctx = (float*)ws;  // reuses q+k region (exactly M_*C_*4 bytes) after attn

  // 1) QKV GEMM (+bias, Q pre-scaled) -> f16 q/k/v in [B,H,N,D]
  gemm_k<0><<<dim3((3*C_)/64, (M_ + 63)/64), 256, 0, stream>>>(
      x, qkv_w, q_bias, k_bias, v_bias, qb, kb, vb, nullptr, M_, C_, 3*C_);

  // 2) RoPE on spatial q/k
  rope_k<<<(B_*H_*S_*(D_/2))/256, 256, 0, stream>>>(qb, kb, rope);

  // 3) Flash attention per (b,h,t)
  attn_k<<<B_*H_*T_*9, 256, 0, stream>>>(qb, kb, vb, ob);

  // 4) Combine token + mean-spatial -> ctx f32 [M, C]
  combine_k<<<(B_*N_*C_ + 255)/256, 256, 0, stream>>>(ob, ctx);

  // 5) Projection GEMM (+proj_b) -> f32 out
  gemm_k<1><<<dim3(C_/64, (M_ + 63)/64), 256, 0, stream>>>(
      ctx, proj_w, proj_b, nullptr, nullptr, nullptr, nullptr, nullptr, out, M_, C_, C_);
}
